// BirthDeathAttention_29841432773256
// MI455X (gfx1250) — matrix operand order repaired and checked by reading, not compiled, not run on hardware
//
#include <hip/hip_runtime.h>
#include <hip/hip_bf16.h>

// ---------------------------------------------------------------------------
// Types
// ---------------------------------------------------------------------------
typedef __bf16 bf16_t;
typedef __attribute__((ext_vector_type(16))) __bf16 bf16x16;
typedef __attribute__((ext_vector_type(8)))  __bf16 bf16x8;
typedef __attribute__((ext_vector_type(4)))  __bf16 bf16x4;
typedef __attribute__((ext_vector_type(8)))  float  f32x8;

#define DIM      1024
#define HEADS    16
#define HEAD_DIM 64
#define SEQ      2048
#define BATCH    2
#define MTOT     (BATCH * SEQ)     // 4096
#define SCALE    0.125f            // 64^-0.5

__device__ __forceinline__ bf16x16 frag_cat(bf16x8 lo, bf16x8 hi) {
  return __builtin_shufflevector(lo, hi, 0,1,2,3,4,5,6,7,8,9,10,11,12,13,14,15);
}

__device__ __forceinline__ f32x8 wmma_bf16(bf16x16 a, bf16x16 b, f32x8 c) {
  // D = A(16x32) * B(32x16) + C, f32 accumulate
  return __builtin_amdgcn_wmma_f32_16x16x32_bf16(false, a, false, b,
                                                 (short)0, c, false, false);
}

// Raw 32-bit LDS byte offset of a shared-memory pointer (generic -> AS3)
__device__ __forceinline__ unsigned lds_off(const void* p) {
  return (unsigned)(unsigned long long)(const __attribute__((address_space(3))) void*)p;
}

// Async global -> LDS 16B copy (per-lane addresses), tracked by ASYNCcnt
__device__ __forceinline__ void async_copy_b128(unsigned lds_byte_off,
                                                const void* gaddr) {
  asm volatile("global_load_async_to_lds_b128 %0, %1, off"
               :: "v"(lds_byte_off), "v"((unsigned long long)gaddr)
               : "memory");
}

__device__ __forceinline__ void wait_async0() {
  asm volatile("s_wait_asynccnt 0x0" ::: "memory");
}

// ---------------------------------------------------------------------------
// fp32 -> bf16 conversion (vectorized by 4)
// ---------------------------------------------------------------------------
__global__ void cvt_f32_bf16(const float4* __restrict__ src,
                             bf16_t* __restrict__ dst, int n4) {
  int i = blockIdx.x * 256 + threadIdx.x;
  if (i < n4) {
    float4 v = src[i];
    bf16x4 o;
    o[0] = (bf16_t)v.x; o[1] = (bf16_t)v.y;
    o[2] = (bf16_t)v.z; o[3] = (bf16_t)v.w;
    *(bf16x4*)(dst + (size_t)i * 4) = o;
  }
}

// ---------------------------------------------------------------------------
// fp32 [rows][cols] -> bf16 transposed [cols][rows] (32x32 LDS tiles)
// ---------------------------------------------------------------------------
__global__ __launch_bounds__(256)
void cvt_transpose_bf16(const float* __restrict__ src, bf16_t* __restrict__ dst,
                        int rows, int cols) {
  __shared__ float t[32][33];
  int r0 = blockIdx.y * 32, c0 = blockIdx.x * 32;
  int tx = threadIdx.x & 31, ty = threadIdx.x >> 5;   // ty: 0..7
#pragma unroll
  for (int i = ty; i < 32; i += 8)
    t[i][tx] = src[(size_t)(r0 + i) * cols + c0 + tx];
  __syncthreads();
#pragma unroll
  for (int i = ty; i < 32; i += 8)
    dst[(size_t)(c0 + i) * rows + r0 + tx] = (bf16_t)t[tx][i];
}

// ---------------------------------------------------------------------------
// Shared WMMA GEMM core: C(128x64 block) = A[M x K] @ Wt^T  (Wt is [N][K])
// 8 waves (4M x 2N), 32x32 per wave, K-step 32, double-buffered LDS filled
// by GLOBAL_LOAD_ASYNC_TO_LDS_B128.
// ---------------------------------------------------------------------------
struct GemmAcc { f32x8 acc[2][2]; };

__device__ __forceinline__ GemmAcc gemm_core(const bf16_t* __restrict__ A,
                                             const bf16_t* __restrict__ Wt,
                                             int K, int m0, int n0,
                                             bf16_t (*lA)[128][40],
                                             bf16_t (*lB)[64][40]) {
  const int tid  = threadIdx.x;
  const int lane = tid & 31;
  const int wave = tid >> 5;
  const int wm = (wave >> 1) * 32;
  const int wn = (wave & 1) * 32;
  const int hiHalf = (lane >= 16);

  // Per-thread async-copy assignments (fixed across K loop)
  const int am = tid >> 1, ahalf = tid & 1;               // A: 256 chunks
  const bf16_t* aSrc = A + (size_t)(m0 + am) * K + ahalf * 16;
  unsigned aDst0 = lds_off(&lA[0][am][ahalf * 16]);
  unsigned aDst1 = lds_off(&lA[1][am][ahalf * 16]);

  const int bn = tid >> 1, bhalf = tid & 1;               // B: 128 chunks
  const bf16_t* bSrc = Wt + (size_t)(n0 + bn) * K + bhalf * 16;
  unsigned bDst0 = lds_off(&lB[0][bn][bhalf * 16]);
  unsigned bDst1 = lds_off(&lB[1][bn][bhalf * 16]);
  const bool doB = (tid < 128);                            // waves 0..3 (uniform)

  // Prologue: fill buffer 0
  async_copy_b128(aDst0, aSrc);
  if (doB) async_copy_b128(bDst0, bSrc);

  GemmAcc g;
#pragma unroll
  for (int i = 0; i < 2; ++i)
#pragma unroll
    for (int j = 0; j < 2; ++j) g.acc[i][j] = (f32x8){};

  int cur = 0;
  for (int k0 = 0; k0 < K; k0 += 32) {
    wait_async0();          // my async copies for buf[cur] complete
    __syncthreads();        // everyone's copies complete

    if (k0 + 32 < K) {      // issue next tile into the other buffer
      unsigned ad = cur ? aDst0 : aDst1;
      async_copy_b128(ad, aSrc + k0 + 32);
      if (doB) {
        unsigned bd = cur ? bDst0 : bDst1;
        async_copy_b128(bd, bSrc + k0 + 32);
      }
    }

    bf16x16 afrag[2], bfrag[2];
    const int ka  = hiHalf ? 16 : 0;  // A: k in {ka..ka+15}
    const int kbb = hiHalf ? 16 : 0;  // B: k in {kbb..kbb+15}
#pragma unroll
    for (int s = 0; s < 2; ++s) {
      int row = wm + s * 16 + (lane & 15);
      afrag[s] = frag_cat(*(const bf16x8*)&lA[cur][row][ka],
                          *(const bf16x8*)&lA[cur][row][ka + 8]);
    }
#pragma unroll
    for (int s = 0; s < 2; ++s) {
      int col = wn + s * 16 + (lane & 15);
      bfrag[s] = frag_cat(*(const bf16x8*)&lB[cur][col][kbb],
                          *(const bf16x8*)&lB[cur][col][kbb + 8]);
    }
#pragma unroll
    for (int i = 0; i < 2; ++i)
#pragma unroll
      for (int j = 0; j < 2; ++j)
        g.acc[i][j] = wmma_bf16(afrag[i], bfrag[j], g.acc[i][j]);

    __syncthreads();        // done reading buf[cur] before it's refilled
    cur ^= 1;
  }
  return g;
}

// ---------------------------------------------------------------------------
// QKV GEMM: X[4096x1024] @ Wqkv -> Q,K [B,H,N,64], V^T [B,H,64,N]
// ---------------------------------------------------------------------------
__global__ __launch_bounds__(256)
void qkv_gemm(const bf16_t* __restrict__ X, const bf16_t* __restrict__ Wt,
              bf16_t* __restrict__ Qb, bf16_t* __restrict__ Kb,
              bf16_t* __restrict__ Vt) {
  __shared__ __attribute__((aligned(16))) bf16_t lA[2][128][40];
  __shared__ __attribute__((aligned(16))) bf16_t lB[2][64][40];

  const int lane = threadIdx.x & 31;
  const int wave = threadIdx.x >> 5;
  const int m0 = blockIdx.x * 128;
  const int n0 = blockIdx.y * 64;
  const int wm = (wave >> 1) * 32;
  const int wn = (wave & 1) * 32;
  const int hiHalf = (lane >= 16);

  GemmAcc g = gemm_core(X, Wt, DIM, m0, n0, lA, lB);

  // Whole 64-wide column tile lies in one matrix & head (both 64-aligned)
  const int mat  = n0 >> 10;            // 0=Q 1=K 2=V
  const int head = (n0 & 1023) >> 6;
#pragma unroll
  for (int i = 0; i < 2; ++i) {
#pragma unroll
    for (int j = 0; j < 2; ++j) {
      int hdi = (wn + j * 16 + (lane & 15)) & 63;
#pragma unroll
      for (int r = 0; r < 8; ++r) {
        int gm = m0 + wm + i * 16 + r + (hiHalf ? 8 : 0);
        int b  = gm >> 11;
        int n  = gm & (SEQ - 1);
        bf16_t val = (bf16_t)g.acc[i][j][r];
        size_t bh = (size_t)(b * HEADS + head);
        if (mat == 0)      Qb[(bh * SEQ + n) * HEAD_DIM + hdi] = val;
        else if (mat == 1) Kb[(bh * SEQ + n) * HEAD_DIM + hdi] = val;
        else               Vt[(bh * HEAD_DIM + hdi) * SEQ + n] = val;
      }
    }
  }
}

// ---------------------------------------------------------------------------
// Projection GEMM: A[4096x1024] @ Wproj + bias -> f32 out
// ---------------------------------------------------------------------------
__global__ __launch_bounds__(256)
void proj_gemm(const bf16_t* __restrict__ A, const bf16_t* __restrict__ Wt,
               const float* __restrict__ bias, float* __restrict__ out) {
  __shared__ __attribute__((aligned(16))) bf16_t lA[2][128][40];
  __shared__ __attribute__((aligned(16))) bf16_t lB[2][64][40];

  const int lane = threadIdx.x & 31;
  const int wave = threadIdx.x >> 5;
  const int m0 = blockIdx.x * 128;
  const int n0 = blockIdx.y * 64;
  const int wm = (wave >> 1) * 32;
  const int wn = (wave & 1) * 32;
  const int hiHalf = (lane >= 16);

  GemmAcc g = gemm_core(A, Wt, DIM, m0, n0, lA, lB);

#pragma unroll
  for (int i = 0; i < 2; ++i) {
#pragma unroll
    for (int j = 0; j < 2; ++j) {
      int col = n0 + wn + j * 16 + (lane & 15);
      float bv = bias[col];
#pragma unroll
      for (int r = 0; r < 8; ++r) {
        int gm = m0 + wm + i * 16 + r + (hiHalf ? 8 : 0);
        out[(size_t)gm * DIM + col] = g.acc[i][j][r] + bv;
      }
    }
  }
}

// ---------------------------------------------------------------------------
// Flash attention: one (b,h) per blockIdx.y, 64 query rows per block
// (4 waves x 16 rows). Online softmax over 32-key blocks.
// ---------------------------------------------------------------------------
__global__ __launch_bounds__(128)
void attn_kernel(const bf16_t* __restrict__ Qb, const bf16_t* __restrict__ Kb,
                 const bf16_t* __restrict__ Vt, const float* __restrict__ imp,
                 const float* __restrict__ pbias, bf16_t* __restrict__ Ob) {
  __shared__ __attribute__((aligned(16))) bf16_t lP[4][16][40]; // P transpose staging

  const int lane = threadIdx.x & 31;
  const int wave = threadIdx.x >> 5;
  const int bh   = blockIdx.y;            // b*HEADS + h
  const int b    = bh >> 4, h = bh & 15;
  const int q0   = blockIdx.x * 64 + wave * 16;
  const int hiHalf = (lane >= 16);

  const bf16_t* Qp = Qb + (size_t)bh * SEQ * HEAD_DIM;
  const bf16_t* Kp = Kb + (size_t)bh * SEQ * HEAD_DIM;
  const bf16_t* Vp = Vt + (size_t)bh * HEAD_DIM * SEQ;

  // per-row additive constants (cancel in softmax; applied for fidelity)
  float rowc[8];
  {
    float pb = pbias[h];
    int rbase = q0 + (hiHalf ? 8 : 0);
#pragma unroll
    for (int r = 0; r < 8; ++r)
      rowc[r] = pb + 0.1f * imp[b * SEQ + rbase + r];
  }

  // Q fragments for hd k-steps 0..31 and 32..63 (A layout: row = lane%16)
  bf16x16 qfrag[2];
  {
    const bf16_t* qr = Qp + (size_t)(q0 + (lane & 15)) * HEAD_DIM;
    int ka = hiHalf ? 16 : 0;
    qfrag[0] = frag_cat(*(const bf16x8*)(qr + ka),
                        *(const bf16x8*)(qr + ka + 8));
    qfrag[1] = frag_cat(*(const bf16x8*)(qr + 32 + ka),
                        *(const bf16x8*)(qr + 32 + ka + 8));
  }

  float m[8], l[8];
#pragma unroll
  for (int r = 0; r < 8; ++r) { m[r] = -1e30f; l[r] = 0.0f; }
  f32x8 o[4] = {};

  for (int kb = 0; kb < SEQ; kb += 32) {
    // prefetch next key/value block (speculative; safe to overrun)
    __builtin_prefetch(Kp + (size_t)(kb + 32) * HEAD_DIM + lane * 4, 0, 1);
    __builtin_prefetch(Vp + (size_t)lane * SEQ + kb + 32, 0, 1);

    // --- S = Q K^T : two 16-key subtiles, 2 k-steps over hd ---
    f32x8 s[2] = {};
#pragma unroll
    for (int t = 0; t < 2; ++t) {
      const bf16_t* kr = Kp + (size_t)(kb + t * 16 + (lane & 15)) * HEAD_DIM;
      int kc = hiHalf ? 16 : 0;
#pragma unroll
      for (int ks = 0; ks < 2; ++ks) {
        bf16x16 bf = frag_cat(*(const bf16x8*)(kr + ks * 32 + kc),
                              *(const bf16x8*)(kr + ks * 32 + kc + 8));
        s[t] = wmma_bf16(qfrag[ks], bf, s[t]);
      }
    }

    // --- online softmax (rows live across 16-lane halves in C layout) ---
#pragma unroll
    for (int r = 0; r < 8; ++r) {
      float v0 = s[0][r] * SCALE + rowc[r];
      float v1 = s[1][r] * SCALE + rowc[r];
      float mx = fmaxf(v0, v1);
#pragma unroll
      for (int off = 8; off >= 1; off >>= 1)
        mx = fmaxf(mx, __shfl_xor(mx, off, 32));
      float mn    = fmaxf(m[r], mx);
      float alpha = __expf(m[r] - mn);
      float p0 = __expf(v0 - mn);
      float p1 = __expf(v1 - mn);
      s[0][r] = p0; s[1][r] = p1;
      float ps = p0 + p1;
#pragma unroll
      for (int off = 8; off >= 1; off >>= 1)
        ps += __shfl_xor(ps, off, 32);
      l[r] = l[r] * alpha + ps;
      m[r] = mn;
#pragma unroll
      for (int t = 0; t < 4; ++t) o[t][r] = o[t][r] * alpha;
    }

    // --- transpose P (C layout) -> A layout via per-wave LDS tile ---
#pragma unroll
    for (int t = 0; t < 2; ++t)
#pragma unroll
      for (int r = 0; r < 8; ++r)
        lP[wave][r + (hiHalf ? 8 : 0)][t * 16 + (lane & 15)] = (bf16_t)s[t][r];
    __asm__ volatile("s_wait_dscnt 0x0" ::: "memory");

    bf16x16 pfrag;
    {
      int row = lane & 15;
      int kc  = hiHalf ? 16 : 0;
      pfrag = frag_cat(*(const bf16x8*)&lP[wave][row][kc],
                       *(const bf16x8*)&lP[wave][row][kc + 8]);
    }
    __asm__ volatile("" ::: "memory");

    // --- O += P V : V^T layout makes B-frags contiguous ---
#pragma unroll
    for (int t = 0; t < 4; ++t) {
      const bf16_t* vr = Vp + (size_t)(t * 16 + (lane & 15)) * SEQ + kb
                       + (hiHalf ? 16 : 0);
      bf16x16 vfrag = frag_cat(*(const bf16x8*)vr, *(const bf16x8*)(vr + 8));
      o[t] = wmma_bf16(pfrag, vfrag, o[t]);
    }
  }

  // Epilogue: O / l -> bf16 attn output [B,N,DIM]
#pragma unroll
  for (int r = 0; r < 8; ++r) {
    float inv = 1.0f / l[r];
    int row = q0 + r + (hiHalf ? 8 : 0);
    size_t orow = ((size_t)b * SEQ + row) * DIM + h * HEAD_DIM;
#pragma unroll
    for (int t = 0; t < 4; ++t)
      Ob[orow + t * 16 + (lane & 15)] = (bf16_t)(o[t][r] * inv);
  }
}

// ---------------------------------------------------------------------------
// Launcher
// ---------------------------------------------------------------------------
extern "C" void kernel_launch(void* const* d_in, const int* in_sizes, int n_in,
                              void* d_out, int out_size, void* d_ws,
                              size_t ws_size, hipStream_t stream) {
  (void)in_sizes; (void)n_in; (void)out_size; (void)ws_size;

  const float* x     = (const float*)d_in[0];
  const float* imp   = (const float*)d_in[1];
  const float* Wqkv  = (const float*)d_in[2];
  const float* Wproj = (const float*)d_in[3];
  const float* bproj = (const float*)d_in[4];
  const float* pbias = (const float*)d_in[5];
  float* out = (float*)d_out;

  char* ws = (char*)d_ws;
  bf16_t* xb   = (bf16_t*)ws;  ws += (size_t)MTOT * DIM * 2;           // 8 MB
  bf16_t* wqbT = (bf16_t*)ws;  ws += (size_t)DIM * 3 * DIM * 2;        // 6 MB  [3072][1024]
  bf16_t* wpbT = (bf16_t*)ws;  ws += (size_t)DIM * DIM * 2;            // 2 MB  [1024][1024]
  bf16_t* Qb   = (bf16_t*)ws;  ws += (size_t)MTOT * DIM * 2;           // 8 MB
  bf16_t* Kb   = (bf16_t*)ws;  ws += (size_t)MTOT * DIM * 2;           // 8 MB
  bf16_t* Vt   = (bf16_t*)ws;  ws += (size_t)MTOT * DIM * 2;           // 8 MB
  bf16_t* Ob   = (bf16_t*)ws;  ws += (size_t)MTOT * DIM * 2;           // 8 MB

  // fp32 -> bf16 activation conversion
  {
    int n4 = MTOT * DIM / 4;
    cvt_f32_bf16<<<(n4 + 255) / 256, 256, 0, stream>>>((const float4*)x, xb, n4);
  }
  // fp32 [K][N] -> bf16 [N][K] transposed weights (contiguous GEMM B tiles)
  cvt_transpose_bf16<<<dim3(3 * DIM / 32, DIM / 32), 256, 0, stream>>>(
      Wqkv, wqbT, DIM, 3 * DIM);
  cvt_transpose_bf16<<<dim3(DIM / 32, DIM / 32), 256, 0, stream>>>(
      Wproj, wpbT, DIM, DIM);

  // QKV projection
  qkv_gemm<<<dim3(MTOT / 128, 3 * DIM / 64), 256, 0, stream>>>(xb, wqbT, Qb, Kb, Vt);

  // Flash attention: 32 q-tiles x 32 (b,h) pairs
  attn_kernel<<<dim3(SEQ / 64, BATCH * HEADS), 128, 0, stream>>>(Qb, Kb, Vt, imp,
                                                                 pbias, Ob);

  // Output projection
  proj_gemm<<<dim3(MTOT / 128, DIM / 64), 256, 0, stream>>>(Ob, wpbT, bproj, out);
}
